// CustomMPNNPredictor_83880711291465
// MI455X (gfx1250) — matrix operand order repaired and checked by reading, not compiled, not run on hardware
//
#include <hip/hip_runtime.h>
#include <cstdint>
#include <cstddef>

// ---------------- problem constants ----------------
#define V_N     30000
#define E_N     60000
#define G_N     1200
#define D_N     64
#define NODEIN  74
#define EH_N    128
#define N_MP    6
#define N_S2S   6
// V/16 = 1875, E/16 = 3750, G/16 = 75 (all exact)

typedef __attribute__((ext_vector_type(16))) __bf16 v16bf;
typedef __attribute__((ext_vector_type(8)))  __bf16 v8bf;
typedef __attribute__((ext_vector_type(8)))  float  v8f;

#define WMMA_BF16(A, B, C) \
  __builtin_amdgcn_wmma_f32_16x16x32_bf16(false, (A), false, (B), (short)0, (C), false, false)

__device__ __forceinline__ float sigmoidf_(float x) { return 1.0f / (1.0f + __expf(-x)); }
__device__ __forceinline__ float reluf_(float x) { return x > 0.0f ? x : 0.0f; }

// A (16x32 bf16) element->k pattern (ISA 7.12.2): lane<16 row=lane, K {0..7,16..23};
// lane>=16 row=lane-16, K {8..15,24..31}
__device__ __forceinline__ int a_frag_k(int e, int grp) {
  int p = e >> 1, w = e & 1;
  return ((p >= 4) ? 16 : 0) + 8 * grp + 2 * (p & 3) + w;
}

// scalar-gather A frag from fp32 row-major matrix
__device__ __forceinline__ v16bf load_a_frag(const float* A, int lda, int row0, int k0, int kdim) {
  int lane = threadIdx.x & 31;
  int r = row0 + (lane & 15);
  int grp = lane >> 4;
  v16bf f;
#pragma unroll
  for (int e = 0; e < 16; ++e) {
    int k = k0 + a_frag_k(e, grp);
    float v = (k < kdim) ? A[(size_t)r * lda + k] : 0.0f;
    f[e] = (__bf16)v;
  }
  return f;
}

// A frag built from x = relu(agg + conv_b), lda=64
__device__ __forceinline__ v16bf load_x_frag(const float* agg, const float* convb, int row0, int k0) {
  int lane = threadIdx.x & 31;
  int r = row0 + (lane & 15);
  int grp = lane >> 4;
  v16bf f;
#pragma unroll
  for (int e = 0; e < 16; ++e) {
    int k = k0 + a_frag_k(e, grp);
    f[e] = (__bf16)reluf_(agg[(size_t)r * 64 + k] + convb[k]);
  }
  return f;
}

// load prepacked B frag: one 32B vector load per lane
__device__ __forceinline__ v16bf load_b_packed(const __bf16* Bp, int nt, int ks, int nks) {
  return ((const v16bf*)Bp)[((size_t)(nt * nks + ks)) * 32 + (threadIdx.x & 31)];
}

// ---------------- kernels ----------------

__global__ void zero_f32_kernel(float* p, size_t n) {
  size_t i = (size_t)blockIdx.x * blockDim.x + threadIdx.x;
  if (i < n) p[i] = 0.0f;
}

// Pre-swizzle a fp32 (K x N) weight matrix into bf16 B-frag layout.
// frag (nt,ks): lane holds 16 bf16 with n = nt*16 + lane%16, k = ks*32 + (e>=8 ? 16 : 0) + 8*(lane/16) + (e&7)
__global__ void pack_b_kernel(const float* __restrict__ W, int ldb, int kdim, int nks, int nwaves,
                              __bf16* __restrict__ out) {
  int wave = blockIdx.x * (blockDim.x >> 5) + (threadIdx.x >> 5);
  if (wave >= nwaves) return;
  int ks = wave % nks;
  int lane = threadIdx.x & 31;
  int n = (wave / nks) * 16 + (lane & 15);
  int grp = lane >> 4;
  v16bf f;
#pragma unroll
  for (int e = 0; e < 16; ++e) {
    int k = ks * 32 + ((e >= 8) ? 16 : 0) + 8 * grp + (e & 7);
    float v = (k < kdim) ? W[(size_t)k * ldb + n] : 0.0f;
    f[e] = (__bf16)v;
  }
  ((v16bf*)out)[(size_t)wave * 32 + lane] = f;
}

// K1: h = relu(node_feats @ proj_W + proj_b)   (V x 74)@(74 x 64)
__global__ void node_proj_kernel(const float* __restrict__ nf, const __bf16* __restrict__ Wp,
                                 const float* __restrict__ b, float* __restrict__ h) {
  int wave = blockIdx.x * (blockDim.x >> 5) + (threadIdx.x >> 5);
  if (wave >= V_N / 16) return;
  int row0 = wave * 16;
  v16bf a0 = load_a_frag(nf, NODEIN, row0, 0, NODEIN);
  v16bf a1 = load_a_frag(nf, NODEIN, row0, 32, NODEIN);
  v16bf a2 = load_a_frag(nf, NODEIN, row0, 64, NODEIN);
  int lane = threadIdx.x & 31, nlo = lane & 15, grp = lane >> 4;
#pragma unroll
  for (int nt = 0; nt < 4; ++nt) {
    v8f acc = {};
    acc = WMMA_BF16(a0, load_b_packed(Wp, nt, 0, 3), acc);
    acc = WMMA_BF16(a1, load_b_packed(Wp, nt, 1, 3), acc);
    acc = WMMA_BF16(a2, load_b_packed(Wp, nt, 2, 3), acc);
    float bb = b[nt * 16 + nlo];
#pragma unroll
    for (int j = 0; j < 8; ++j) {
      int m = row0 + j + 8 * grp;
      h[(size_t)m * 64 + nt * 16 + nlo] = reluf_(acc[j] + bb);
    }
  }
}

// K2: he = relu(edge_feats @ eW1 + eb1), written DIRECTLY in packed A-frag bf16 layout.
// wave = mtile*4 + ks; lane holds 16 values for row r = mtile*16 + lane%16.
__global__ void edge_hidden_pack_kernel(const float* __restrict__ ef, const float* __restrict__ W1,
                                        const float* __restrict__ b1, __bf16* __restrict__ hep) {
  int wave = blockIdx.x * 8 + (threadIdx.x >> 5);
  if (wave >= (E_N / 16) * 4) return;
  int ks = wave & 3;
  int lane = threadIdx.x & 31, grp = lane >> 4;
  int r = (wave >> 2) * 16 + (lane & 15);
  float er[12];
#pragma unroll
  for (int j = 0; j < 12; ++j) er[j] = ef[(size_t)r * 12 + j];
  v16bf f;
#pragma unroll
  for (int e = 0; e < 16; ++e) {
    int k = ks * 32 + a_frag_k(e, grp);
    float a = b1[k];
#pragma unroll
    for (int j = 0; j < 12; ++j) a += er[j] * W1[j * EH_N + k];
    f[e] = (__bf16)reluf_(a);
  }
  ((v16bf*)hep)[(size_t)wave * 32 + lane] = f;
}

// K3: We = he @ eW2 + eb2  (E x 128)@(128 x 4096), bf16 output in tile-swizzled layout:
// We_s[mtile][c][r] (r = row-in-tile, innermost) -> lane stores 8 contiguous bf16 (b128).
__global__ void we_gemm_kernel(const __bf16* __restrict__ hep, const __bf16* __restrict__ W2p,
                               const float* __restrict__ b2, __bf16* __restrict__ We_s) {
  int wave = blockIdx.x * 8 + (threadIdx.x >> 5);
  if (wave >= (E_N / 16) * 32) return;
  int mtile = wave >> 5;  // 0..3749
  int ngrp = wave & 31;   // covers 8 N-tiles
  int lane = threadIdx.x & 31, nlo = lane & 15, grp = lane >> 4;
  v16bf a[4];
#pragma unroll
  for (int ks = 0; ks < 4; ++ks)
    a[ks] = ((const v16bf*)hep)[((size_t)mtile * 4 + ks) * 32 + lane];
  for (int t = 0; t < 8; ++t) {
    int nt = ngrp * 8 + t;
    v8f acc = {};
#pragma unroll
    for (int ks = 0; ks < 4; ++ks)
      acc = WMMA_BF16(a[ks], load_b_packed(W2p, nt, ks, 4), acc);
    int c = nt * 16 + nlo;
    float bb = b2[c];
    v8bf o8;
#pragma unroll
    for (int j = 0; j < 8; ++j) o8[j] = (__bf16)(acc[j] + bb);
    // element offset = (mtile*4096 + c)*16 + 8*grp ; in v8bf units: *2 + grp
    ((v8bf*)We_s)[((size_t)mtile * 4096 + c) * 2 + grp] = o8;
  }
}

// K4: per-edge messages from swizzled We_s, scattered into agg[dst].
// Block = 256 threads per edge-tile (16 edges). h rows staged in LDS via async loads.
__global__ void message_kernel(const float* __restrict__ h, const __bf16* __restrict__ We_s,
                               const int* __restrict__ src, const int* __restrict__ dst,
                               float* __restrict__ agg) {
  int mt = blockIdx.x;          // 0..3749
  int t = threadIdx.x;          // 0..255
  __shared__ float hs_lds[16][68];   // stride 68 floats: 16B-aligned rows, conflict-free
  __shared__ int sidx[16], didx[16];
  if (t < 16) {
    sidx[t] = src[mt * 16 + t];
    didx[t] = dst[mt * 16 + t];
  }
  __syncthreads();
  // async stage: thread t copies 16B of h[src[r]] into LDS
  {
    int r = t >> 4;
    int i0 = (t & 15) * 4;
    const float* gp = h + (size_t)sidx[r] * 64 + i0;
    unsigned lds = (unsigned)(size_t)(void*)&hs_lds[r][i0];
    asm volatile("global_load_async_to_lds_b128 %0, %1, off"
                 :: "v"(lds), "v"((unsigned long long)(size_t)gp) : "memory");
  }
  asm volatile("s_wait_asynccnt 0x0" ::: "memory");
  __syncthreads();

  int rp = t & 7;       // r pair: rows 2rp, 2rp+1
  int o = t >> 3;       // 0..31 ; second half of o via +32
  const unsigned* Wp = (const unsigned*)We_s + (size_t)mt * 4096 * 8;  // tile block (uint units)
  float a00 = 0.f, a01 = 0.f, a10 = 0.f, a11 = 0.f;
#pragma unroll 4
  for (int i = 0; i < 64; ++i) {
    float h0 = hs_lds[2 * rp][i];
    float h1 = hs_lds[2 * rp + 1][i];
    unsigned p0 = Wp[(size_t)(i * 64 + o) * 8 + rp];
    unsigned p1 = Wp[(size_t)(i * 64 + o + 32) * 8 + rp];
    a00 += h0 * __uint_as_float(p0 << 16);
    a01 += h1 * __uint_as_float(p0 & 0xffff0000u);
    a10 += h0 * __uint_as_float(p1 << 16);
    a11 += h1 * __uint_as_float(p1 & 0xffff0000u);
  }
  atomicAdd(&agg[(size_t)didx[2 * rp] * 64 + o], a00);
  atomicAdd(&agg[(size_t)didx[2 * rp + 1] * 64 + o], a01);
  atomicAdd(&agg[(size_t)didx[2 * rp] * 64 + o + 32], a10);
  atomicAdd(&agg[(size_t)didx[2 * rp + 1] * 64 + o + 32], a11);
}

// K5: fused GRU with WMMA gate GEMMs (packed B).
__global__ void gru_kernel(const float* __restrict__ agg, const float* __restrict__ convb,
                           const float* __restrict__ hidden,
                           const __bf16* __restrict__ Wihp, const float* __restrict__ bih,
                           const __bf16* __restrict__ Whhp, const float* __restrict__ bhh,
                           float* __restrict__ outh) {
  int wave = blockIdx.x * 8 + (threadIdx.x >> 5);
  if (wave >= V_N / 16) return;
  int row0 = wave * 16;
  v16bf xa[2], ha[2];
#pragma unroll
  for (int ks = 0; ks < 2; ++ks) {
    xa[ks] = load_x_frag(agg, convb, row0, ks * 32);
    ha[ks] = load_a_frag(hidden, 64, row0, ks * 32, 64);
  }
  int lane = threadIdx.x & 31, nlo = lane & 15, grp = lane >> 4;
#pragma unroll
  for (int nt = 0; nt < 4; ++nt) {
    v8f ir = {}, iz = {}, in_ = {}, hr = {}, hz = {}, hn = {};
#pragma unroll
    for (int ks = 0; ks < 2; ++ks) {
      ir  = WMMA_BF16(xa[ks], load_b_packed(Wihp, 0 + nt, ks, 2), ir);
      iz  = WMMA_BF16(xa[ks], load_b_packed(Wihp, 4 + nt, ks, 2), iz);
      in_ = WMMA_BF16(xa[ks], load_b_packed(Wihp, 8 + nt, ks, 2), in_);
      hr  = WMMA_BF16(ha[ks], load_b_packed(Whhp, 0 + nt, ks, 2), hr);
      hz  = WMMA_BF16(ha[ks], load_b_packed(Whhp, 4 + nt, ks, 2), hz);
      hn  = WMMA_BF16(ha[ks], load_b_packed(Whhp, 8 + nt, ks, 2), hn);
    }
    int col = nt * 16 + nlo;
    float bir = bih[col], biz = bih[64 + col], bin = bih[128 + col];
    float bhr = bhh[col], bhz = bhh[64 + col], bhn = bhh[128 + col];
#pragma unroll
    for (int j = 0; j < 8; ++j) {
      int m = row0 + j + 8 * grp;
      float hv = hidden[(size_t)m * 64 + col];
      float r = sigmoidf_(ir[j] + bir + hr[j] + bhr);
      float z = sigmoidf_(iz[j] + biz + hz[j] + bhz);
      float nn = tanhf(in_[j] + bin + r * (hn[j] + bhn));
      outh[(size_t)m * 64 + col] = (1.0f - z) * nn + z * hv;
    }
  }
}

// Graph node-range discovery (gids sorted)
__global__ void bounds_init_kernel(int* gstart, int* gend) {
  int g = blockIdx.x * 256 + threadIdx.x;
  if (g < G_N) { gstart[g] = V_N; gend[g] = 0; }
}
__global__ void bounds_kernel(const int* __restrict__ gids, int* gstart, int* gend) {
  int v = blockIdx.x * 256 + threadIdx.x;
  if (v >= V_N) return;
  int g = gids[v];
  atomicMin(&gstart[g], v);
  atomicMax(&gend[g], v + 1);
}

// Set2Set 3-layer LSTM stack: one block (64 threads) per graph
__global__ void s2s_lstm_kernel(const float* __restrict__ qstar,
    const float* Wih0, const float* bih0, const float* Whh0, const float* bhh0,
    const float* Wih1, const float* bih1, const float* Whh1, const float* bhh1,
    const float* Wih2, const float* bih2, const float* Whh2, const float* bhh2,
    float* __restrict__ hs, float* __restrict__ cs, float* __restrict__ q) {
  int g = blockIdx.x;
  int d = threadIdx.x;  // 0..63
  __shared__ float xs[128];
  __shared__ float hb[64];
  xs[d] = qstar[(size_t)g * 128 + d];
  xs[64 + d] = qstar[(size_t)g * 128 + 64 + d];
  const float* Wihs[3] = {Wih0, Wih1, Wih2};
  const float* bihs[3] = {bih0, bih1, bih2};
  const float* Whhs[3] = {Whh0, Whh1, Whh2};
  const float* bhhs[3] = {bhh0, bhh1, bhh2};
  const int dins[3] = {128, 64, 64};
  for (int l = 0; l < 3; ++l) {
    __syncthreads();
    hb[d] = hs[((size_t)l * G_N + g) * 64 + d];
    __syncthreads();
    const float* Wih = Wihs[l];
    const float* Whh = Whhs[l];
    float gi = bihs[l][d] + bhhs[l][d];
    float gf = bihs[l][64 + d] + bhhs[l][64 + d];
    float gg = bihs[l][128 + d] + bhhs[l][128 + d];
    float go = bihs[l][192 + d] + bhhs[l][192 + d];
    int din = dins[l];
    for (int k = 0; k < din; ++k) {
      float xv = xs[k];
      gi += xv * Wih[k * 256 + d];
      gf += xv * Wih[k * 256 + 64 + d];
      gg += xv * Wih[k * 256 + 128 + d];
      go += xv * Wih[k * 256 + 192 + d];
    }
    for (int k = 0; k < 64; ++k) {
      float hv = hb[k];
      gi += hv * Whh[k * 256 + d];
      gf += hv * Whh[k * 256 + 64 + d];
      gg += hv * Whh[k * 256 + 128 + d];
      go += hv * Whh[k * 256 + 192 + d];
    }
    float cold = cs[((size_t)l * G_N + g) * 64 + d];
    float c = sigmoidf_(gf) * cold + sigmoidf_(gi) * tanhf(gg);
    float hh = sigmoidf_(go) * tanhf(c);
    cs[((size_t)l * G_N + g) * 64 + d] = c;
    hs[((size_t)l * G_N + g) * 64 + d] = hh;
    __syncthreads();
    xs[d] = hh;
  }
  __syncthreads();
  q[(size_t)g * 64 + d] = xs[d];
}

// attention scores e[v] = dot(h[v], q[gids[v]])
__global__ void score_kernel(const float* __restrict__ h, const float* __restrict__ q,
                             const int* __restrict__ gids, float* __restrict__ esc) {
  int v = blockIdx.x * 256 + threadIdx.x;
  if (v >= V_N) return;
  const float* hr = h + (size_t)v * 64;
  const float* qr = q + (size_t)gids[v] * 64;
  float s = 0.0f;
#pragma unroll
  for (int k = 0; k < 64; ++k) s += hr[k] * qr[k];
  esc[v] = s;
}

// per-graph softmax + weighted readout; writes q_star = [q, readout]
__global__ void readout_kernel(const float* __restrict__ h, const float* __restrict__ esc,
                               const int* __restrict__ gstart, const int* __restrict__ gend,
                               const float* __restrict__ q, float* __restrict__ qstar) {
  int g = blockIdx.x;
  int t = threadIdx.x;  // 0..255
  __shared__ float red[256];
  __shared__ float smax, ssum;
  int s0 = gstart[g], s1 = gend[g];
  float mx = -3.0e38f;
  for (int v = s0 + t; v < s1; v += 256) { float e = esc[v]; mx = e > mx ? e : mx; }
  red[t] = mx;
  __syncthreads();
  for (int off = 128; off > 0; off >>= 1) {
    if (t < off) { float o = red[t + off]; if (o > red[t]) red[t] = o; }
    __syncthreads();
  }
  if (t == 0) smax = red[0];
  __syncthreads();
  mx = smax;
  float sm = 0.0f;
  for (int v = s0 + t; v < s1; v += 256) sm += __expf(esc[v] - mx);
  __syncthreads();
  red[t] = sm;
  __syncthreads();
  for (int off = 128; off > 0; off >>= 1) {
    if (t < off) red[t] += red[t + off];
    __syncthreads();
  }
  if (t == 0) ssum = red[0];
  __syncthreads();
  float denom = ssum;
  int d = t & 63, sub = t >> 6;
  float acc = 0.0f;
  for (int v = s0 + sub; v < s1; v += 4) acc += h[(size_t)v * 64 + d] * __expf(esc[v] - mx);
  __syncthreads();
  red[t] = acc;
  __syncthreads();
  if (t < 64) {
    float r = red[t] + red[t + 64] + red[t + 128] + red[t + 192];
    r = denom > 0.0f ? r / denom : 0.0f;
    qstar[(size_t)g * 128 + t] = q[(size_t)g * 64 + t];
    qstar[(size_t)g * 128 + 64 + t] = r;
  }
}

// final process + predict MLPs: block(64) per graph
__global__ void predict_kernel(const float* __restrict__ qstar,
    const float* pW, const float* pb, const float* qW1, const float* qb1,
    const float* qW2, const float* qb2, const float* qW3, const float* qb3,
    float* __restrict__ out) {
  int g = blockIdx.x;
  int d = threadIdx.x;  // 0..63
  __shared__ float s0[128];
  __shared__ float s1[64];
  __shared__ float s2[64];
  s0[d] = qstar[(size_t)g * 128 + d];
  s0[64 + d] = qstar[(size_t)g * 128 + 64 + d];
  __syncthreads();
  float a = pb[d];
  for (int k = 0; k < 128; ++k) a += s0[k] * pW[k * 64 + d];
  s1[d] = reluf_(a);
  __syncthreads();
  float b = qb1[d];
  for (int k = 0; k < 64; ++k) b += s1[k] * qW1[k * 64 + d];
  s2[d] = reluf_(b);
  __syncthreads();
  float c = qb2[d];
  for (int k = 0; k < 64; ++k) c += s2[k] * qW2[k * 64 + d];
  c = reluf_(c);
  __syncthreads();
  s1[d] = c * qW3[d];
  __syncthreads();
  for (int off = 32; off > 0; off >>= 1) {
    if (d < off) s1[d] += s1[d + off];
    __syncthreads();
  }
  if (d == 0) out[g] = s1[0] + qb3[0];
}

// ---------------- host launcher ----------------
extern "C" void kernel_launch(void* const* d_in, const int* in_sizes, int n_in,
                              void* d_out, int out_size, void* d_ws, size_t ws_size,
                              hipStream_t stream) {
  const float* node_feats = (const float*)d_in[0];
  const float* edge_feats = (const float*)d_in[1];
  const int* src = (const int*)d_in[2];
  const int* dst = (const int*)d_in[3];
  const int* gids = (const int*)d_in[4];
  const float* proj_W = (const float*)d_in[5];
  const float* proj_b = (const float*)d_in[6];
  const float* eW1 = (const float*)d_in[7];
  const float* eb1 = (const float*)d_in[8];
  const float* eW2 = (const float*)d_in[9];
  const float* eb2 = (const float*)d_in[10];
  const float* conv_b = (const float*)d_in[11];
  const float* gWih = (const float*)d_in[12];
  const float* gbih = (const float*)d_in[13];
  const float* gWhh = (const float*)d_in[14];
  const float* gbhh = (const float*)d_in[15];
  const float* pW = (const float*)d_in[16];
  const float* pb = (const float*)d_in[17];
  const float* qW1 = (const float*)d_in[18];
  const float* qb1 = (const float*)d_in[19];
  const float* qW2 = (const float*)d_in[20];
  const float* qb2 = (const float*)d_in[21];
  const float* qW3 = (const float*)d_in[22];
  const float* qb3 = (const float*)d_in[23];
  const float* lW[12];
  for (int i = 0; i < 12; ++i) lW[i] = (const float*)d_in[24 + i];
  float* out = (float*)d_out;

  // workspace carve
  char* p = (char*)d_ws;
  auto take = [&](size_t bytes) -> char* {
    char* r = (char*)(((uintptr_t)p + 255) & ~(uintptr_t)255);
    p = r + bytes;
    return r;
  };
  __bf16* We_s = (__bf16*)take((size_t)E_N * 4096 * 2);       // swizzled edge weight tensor
  __bf16* hep  = (__bf16*)take((size_t)(E_N / 16) * 4 * 1024); // packed A frags of he
  __bf16* W2p  = (__bf16*)take((size_t)256 * 4 * 1024);        // packed eW2
  __bf16* projWp = (__bf16*)take((size_t)4 * 3 * 1024);        // packed proj_W
  __bf16* Wihp = (__bf16*)take((size_t)12 * 2 * 1024);         // packed gru_Wih
  __bf16* Whhp = (__bf16*)take((size_t)12 * 2 * 1024);         // packed gru_Whh
  float* hA = (float*)take((size_t)V_N * 64 * 4);
  float* hB = (float*)take((size_t)V_N * 64 * 4);
  float* agg = (float*)take((size_t)V_N * 64 * 4);
  float* esc = (float*)take((size_t)V_N * 4);
  float* qv = (float*)take((size_t)G_N * 64 * 4);
  float* qstar = (float*)take((size_t)G_N * 128 * 4);
  float* hs = (float*)take((size_t)3 * G_N * 64 * 4);
  float* cs = (float*)take((size_t)3 * G_N * 64 * 4);
  int* gstart = (int*)take((size_t)G_N * 4);
  int* gend = (int*)take((size_t)G_N * 4);

  // ---- weight prepack (B frag layouts) ----
  pack_b_kernel<<<dim3(2), dim3(256), 0, stream>>>(proj_W, 64, NODEIN, 3, 12, projWp);
  pack_b_kernel<<<dim3(128), dim3(256), 0, stream>>>(eW2, 4096, 128, 4, 1024, W2p);
  pack_b_kernel<<<dim3(3), dim3(256), 0, stream>>>(gWih, 192, 64, 2, 24, Wihp);
  pack_b_kernel<<<dim3(3), dim3(256), 0, stream>>>(gWhh, 192, 64, 2, 24, Whhp);

  // ---- MPNN ----
  node_proj_kernel<<<dim3((V_N / 16 + 3) / 4), dim3(128), 0, stream>>>(node_feats, projWp, proj_b, hA);
  edge_hidden_pack_kernel<<<dim3((E_N / 16) * 4 / 8), dim3(256), 0, stream>>>(edge_feats, eW1, eb1, hep);
  we_gemm_kernel<<<dim3((E_N / 16) * 32 / 8), dim3(256), 0, stream>>>(hep, W2p, eb2, We_s);

  float* hcur = hA;
  float* hnext = hB;
  for (int s = 0; s < N_MP; ++s) {
    zero_f32_kernel<<<dim3((V_N * 64 + 255) / 256), dim3(256), 0, stream>>>(agg, (size_t)V_N * 64);
    message_kernel<<<dim3(E_N / 16), dim3(256), 0, stream>>>(hcur, We_s, src, dst, agg);
    gru_kernel<<<dim3((V_N / 16 + 7) / 8), dim3(256), 0, stream>>>(agg, conv_b, hcur, Wihp, gbih, Whhp, gbhh, hnext);
    float* t = hcur; hcur = hnext; hnext = t;
  }

  // ---- Set2Set ----
  bounds_init_kernel<<<dim3((G_N + 255) / 256), dim3(256), 0, stream>>>(gstart, gend);
  bounds_kernel<<<dim3((V_N + 255) / 256), dim3(256), 0, stream>>>(gids, gstart, gend);
  zero_f32_kernel<<<dim3((3 * G_N * 64 + 255) / 256), dim3(256), 0, stream>>>(hs, (size_t)3 * G_N * 64);
  zero_f32_kernel<<<dim3((3 * G_N * 64 + 255) / 256), dim3(256), 0, stream>>>(cs, (size_t)3 * G_N * 64);
  zero_f32_kernel<<<dim3((G_N * 128 + 255) / 256), dim3(256), 0, stream>>>(qstar, (size_t)G_N * 128);

  for (int it = 0; it < N_S2S; ++it) {
    s2s_lstm_kernel<<<dim3(G_N), dim3(64), 0, stream>>>(qstar,
        lW[0], lW[1], lW[2], lW[3], lW[4], lW[5], lW[6], lW[7], lW[8], lW[9], lW[10], lW[11],
        hs, cs, qv);
    score_kernel<<<dim3((V_N + 255) / 256), dim3(256), 0, stream>>>(hcur, qv, gids, esc);
    readout_kernel<<<dim3(G_N), dim3(256), 0, stream>>>(hcur, esc, gstart, gend, qv, qstar);
  }

  // ---- predict ----
  predict_kernel<<<dim3(G_N), dim3(64), 0, stream>>>(qstar, pW, pb, qW1, qb1, qW2, qb2, qW3, qb3, out);
}
